// DGCNN_5815385718918
// MI455X (gfx1250) — matrix operand order repaired and checked by reading, not compiled, not run on hardware
//
#include <hip/hip_runtime.h>
#include <hip/hip_bf16.h>

// ---------------------------------------------------------------------------
// DGCNN forward for MI455X (gfx1250, wave32, WMMA + TDM).
// Edge conv decomposition:  y[b,o,n,j] = v[b,o,idx(n,j)] + u[b,o,n]
//   with v = W_diff * x ,  u = (W_ctr - W_diff) * x   (two dense GEMMs).
// BN(axis over B,N,k) + leakyrelu + max_k handled via (max,min) trick since
// BN/lrelu are monotone per channel given sign(scale).
// GEMM B-operand tiles are staged to LDS by the Tensor Data Mover.
// ---------------------------------------------------------------------------

#define KNN_K 20
#define SLOPE 0.2f
#define BN_EPS 1e-5f

typedef __attribute__((ext_vector_type(16))) __bf16 bf16x16;
typedef __attribute__((ext_vector_type(8)))  float  floatx8;
typedef __attribute__((ext_vector_type(4)))  unsigned tdm_v4u;
typedef __attribute__((ext_vector_type(8)))  int      tdm_v8i;
typedef __attribute__((ext_vector_type(4)))  int      tdm_v4i;

union Frag { bf16x16 v; unsigned u[8]; };

__device__ __forceinline__ unsigned short f2bf(float f) {
    unsigned u = __float_as_uint(f);
    u += 0x7fffu + ((u >> 16) & 1u);   // round-to-nearest-even
    return (unsigned short)(u >> 16);
}

// ---------------------------------------------------------------------------
// bf16 WMMA GEMM with TDM-staged activations.
// For each batch b: D[m,n] = sum_k A[m,k] * Bact[b,n,k]
//   A  : bf16 row-major [M][lda]  (weights, K-padded, lda multiple of 32)
//   B  : bf16 point-major, element (b,k,n) at b*bStrideB + n*ldb + c0 + k
//   D  : f32 [b][M][Npts]
// One wave per block; block tile = 16(M) x 64(N); K chunked by 256 through
// LDS via tensor_load_to_lds with +4-dword row padding (bank spread + b128).
// ---------------------------------------------------------------------------
__global__ __launch_bounds__(32)
void gemm_bf16_wmma(const unsigned short* __restrict__ A, int lda,
                    const unsigned short* __restrict__ Bact, long long bStrideB,
                    int ldb, int c0,
                    float* __restrict__ D, long long bStrideD,
                    int M, int Npts, int K)
{
    __shared__ unsigned btile[64 * 132];   // 64 rows x (<=128 + 4 pad) dwords

    int b  = blockIdx.z;
    int n0 = blockIdx.x * 64;
    int m0 = blockIdx.y * 16;
    int l  = threadIdx.x;
    int lo = l & 15, hi = l >> 4;

    const unsigned* Arow = (const unsigned*)(A + (size_t)(m0 + lo) * lda);
    const unsigned short* Bbase =
        Bact + (size_t)b * bStrideB + (size_t)n0 * ldb + c0;

    floatx8 acc[4];
    #pragma unroll
    for (int s = 0; s < 4; ++s) acc[s] = (floatx8){0.f,0.f,0.f,0.f,0.f,0.f,0.f,0.f};

    for (int kc = 0; kc < K; kc += 256) {
        int chunkK = (K - kc < 256) ? (K - kc) : 256;   // power of two here
        int rowD = chunkK >> 1;                          // dwords per tile row
        int rsd  = rowD + 4;                             // padded LDS row stride
        int padCode = (31 - __builtin_clz((unsigned)chunkK)) - 2; // log2-2: interval=rowD dwords

        // ---- Tensor DMA descriptor (ISA ch.8): 2D tile [64 rows][chunkK bf16]
        unsigned ldsA = (unsigned)(size_t)&btile[0];
        unsigned long long ga = (unsigned long long)(size_t)(Bbase + kc);
        tdm_v4u g0 = { 1u,                               // count=1, user desc
                       ldsA,                             // lds_addr
                       (unsigned)ga,                     // global_addr[31:0]
                       (unsigned)((ga >> 32) & 0x1FFFFFFu) | (2u << 30) }; // addr[56:32] | type=2
        unsigned d0 = (1u << 16)                         // data_size = 2 bytes
                    | (1u << 20)                         // pad_enable
                    | ((unsigned)padCode << 22)          // pad_interval = rowD dwords
                    | (3u << 25);                        // pad_amount = 4 dwords
        unsigned td0 = (unsigned)chunkK, td1 = 64u;      // tensor dims (elements)
        unsigned long long s0 = (unsigned long long)ldb; // row stride (elements)
        tdm_v8i g1 = { (int)d0,
                       (int)((td0 & 0xFFFFu) << 16),
                       (int)(((td0 >> 16) & 0xFFFFu) | ((td1 & 0xFFFFu) << 16)),
                       (int)(((td1 >> 16) & 0xFFFFu) | (((unsigned)chunkK & 0xFFFFu) << 16)), // tile_dim0
                       (int)(64u & 0xFFFFu),             // tile_dim1 = 64, tile_dim2 = 0
                       (int)(s0 & 0xFFFFFFFFu),
                       (int)((s0 >> 32) & 0xFFFFu),      // stride0 hi | stride1 lo(=0)
                       0 };
        tdm_v4i gz = { 0, 0, 0, 0 };
#if defined(__clang_major__) && __clang_major__ >= 23
        tdm_v8i gz8 = { 0, 0, 0, 0, 0, 0, 0, 0 };
        __builtin_amdgcn_tensor_load_to_lds(g0, g1, gz, gz, gz8, 0);
#else
        __builtin_amdgcn_tensor_load_to_lds(g0, g1, gz, gz, 0);
#endif
        __builtin_amdgcn_s_wait_tensorcnt(0);   // single wave per block: no barrier

        for (int k0 = 0; k0 < chunkK; k0 += 32) {
            Frag af;
            const uint4* Ap = (const uint4*)(Arow + ((kc + k0) >> 1) + hi * 4);
            uint4 a0 = Ap[0];          // dwords +0..3   (K = {0..7}+hi*8  pairs)
            uint4 a1 = Ap[2];          // dwords +8..11  (K = {16..23}+hi*8 pairs)
            af.u[0] = a0.x; af.u[1] = a0.y; af.u[2] = a0.z; af.u[3] = a0.w;
            af.u[4] = a1.x; af.u[5] = a1.y; af.u[6] = a1.z; af.u[7] = a1.w;

            #pragma unroll
            for (int s = 0; s < 4; ++s) {
                int w = (s * 16 + lo) * rsd + (k0 >> 1) + hi * 4;
                const uint4* bp = (const uint4*)&btile[w];
                uint4 b0 = bp[0], b1 = bp[2];
                Frag bf_;
                bf_.u[0] = b0.x; bf_.u[1] = b0.y; bf_.u[2] = b0.z; bf_.u[3] = b0.w;
                bf_.u[4] = b1.x; bf_.u[5] = b1.y; bf_.u[6] = b1.z; bf_.u[7] = b1.w;
                acc[s] = __builtin_amdgcn_wmma_f32_16x16x32_bf16(
                    false, af.v, false, bf_.v, (short)0, acc[s], false, false);
            }
        }
    }

    #pragma unroll
    for (int s = 0; s < 4; ++s) {
        float* Drow = D + (size_t)b * bStrideD + n0 + s * 16 + lo;
        #pragma unroll
        for (int r = 0; r < 8; ++r) {
            Drow[(size_t)(m0 + r + hi * 8) * Npts] = acc[s][r];  // M = r + 8*hi
        }
    }
}

// ---------------------------------------------------------------------------
// kNN: per (b, point) keep 20 smallest squared distances (ties -> lower idx,
// matching jax top_k). Neighbor tiles streamed to LDS with async-to-LDS ops.
// ---------------------------------------------------------------------------
__global__ __launch_bounds__(64)
void knn_kernel(const float* __restrict__ feat, long long bStride,
                int C, int Npts, int* __restrict__ idx)
{
    __shared__ float cen[128 * 64];   // [c][p] for 64 center points
    __shared__ float nb [128 * 32];   // [c][p] for 32 neighbor points

    int b = blockIdx.y, t = threadIdx.x;
    int n0 = blockIdx.x * 64;
    const float* fb = feat + (size_t)b * bStride;

    for (int e = t; e < C * 64; e += 64)
        cen[e] = fb[(size_t)(e >> 6) * Npts + n0 + (e & 63)];

    float bd[KNN_K]; int bi[KNN_K];
    #pragma unroll
    for (int j = 0; j < KNN_K; ++j) { bd[j] = 3.4e38f; bi[j] = 0; }

    for (int m0 = 0; m0 < Npts; m0 += 32) {
        __syncthreads();   // previous tile fully consumed (also covers cen)
        for (int e = t; e < C * 32; e += 64) {
            unsigned lds_off = (unsigned)(size_t)&nb[e];
            unsigned long long ga =
                (unsigned long long)(size_t)&fb[(size_t)(e >> 5) * Npts + m0 + (e & 31)];
            asm volatile("global_load_async_to_lds_b32 %0, %1, off"
                         :: "v"(lds_off), "v"(ga) : "memory");
        }
        asm volatile("s_wait_asynccnt 0x0" ::: "memory");
        __syncthreads();

        for (int mm = 0; mm < 32; ++mm) {
            float d = 0.f;
            for (int c = 0; c < C; ++c) {
                float df = cen[c * 64 + t] - nb[c * 32 + mm];
                d = fmaf(df, df, d);
            }
            if (d < bd[KNN_K - 1]) {          // insert + single bubble pass
                bd[KNN_K - 1] = d; bi[KNN_K - 1] = m0 + mm;
                #pragma unroll
                for (int j = KNN_K - 1; j > 0; --j) {
                    if (bd[j] < bd[j - 1]) {
                        float td = bd[j]; bd[j] = bd[j-1]; bd[j-1] = td;
                        int   ti = bi[j]; bi[j] = bi[j-1]; bi[j-1] = ti;
                    }
                }
            }
        }
    }
    int* op = idx + ((size_t)b * Npts + n0 + t) * KNN_K;
    #pragma unroll
    for (int j = 0; j < KNN_K; ++j) op[j] = bi[j];
}

// ---------------------------------------------------------------------------
// Gather neighbors, y = v[idx]+u ; track per-(b,o,n) max/min over k and
// accumulate per-channel sum/sumsq (wave-reduce + atomics).
// block (32 pts, 8 channels)
// ---------------------------------------------------------------------------
__global__ __launch_bounds__(256)
void gather_stats_kernel(const float* __restrict__ u, const float* __restrict__ v,
                         const int* __restrict__ idx,
                         float* __restrict__ ymax, float* __restrict__ ymin,
                         float* __restrict__ chS, float* __restrict__ chS2,
                         int O, int Npts)
{
    __shared__ int sidx[32 * KNN_K];
    int tx = threadIdx.x, ty = threadIdx.y;
    int b = blockIdx.z;
    int n = blockIdx.x * 32 + tx;
    int o = blockIdx.y * 8 + ty;
    int tid = ty * 32 + tx;

    for (int e = tid; e < 32 * KNN_K; e += 256)
        sidx[e] = idx[((size_t)b * Npts + blockIdx.x * 32) * KNN_K + e];
    __syncthreads();

    size_t base = ((size_t)b * O + o) * Npts;
    float uc = u[base + n];
    const float* vr = v + base;

    float mx = -3.4e38f, mn = 3.4e38f, s = 0.f, s2 = 0.f;
    #pragma unroll
    for (int j = 0; j < KNN_K; ++j) {
        int m = sidx[tx * KNN_K + j];
        float val = vr[m] + uc;
        mx = fmaxf(mx, val); mn = fminf(mn, val);
        s += val; s2 = fmaf(val, val, s2);
    }
    ymax[base + n] = mx;
    ymin[base + n] = mn;

    #pragma unroll
    for (int off = 16; off > 0; off >>= 1) {
        s  += __shfl_xor(s,  off, 32);
        s2 += __shfl_xor(s2, off, 32);
    }
    if (tx == 0) { atomicAdd(&chS[o], s); atomicAdd(&chS2[o], s2); }
}

__global__ void bn_prep_kernel(const float* __restrict__ chS, const float* __restrict__ chS2,
                               const float* __restrict__ g, const float* __restrict__ bb,
                               float* __restrict__ scaleA, float* __restrict__ shiftA,
                               int O, float invCnt)
{
    int o = blockIdx.x * blockDim.x + threadIdx.x;
    if (o >= O) return;
    float mean = chS[o] * invCnt;
    float var  = chS2[o] * invCnt - mean * mean;
    float sc = g[o] * rsqrtf(var + BN_EPS);
    scaleA[o] = sc;
    shiftA[o] = bb[o] - mean * sc;
}

// pick max/min per sign(scale), BN affine, leaky relu, write f32 + bf16 copies
__global__ __launch_bounds__(256)
void finalize_edge_kernel(const float* __restrict__ ymax, const float* __restrict__ ymin,
                          const float* __restrict__ scaleA, const float* __restrict__ shiftA,
                          float* __restrict__ xc, unsigned short* __restrict__ xcbf,
                          int O, int c0, int Npts)
{
    size_t i = (size_t)blockIdx.x * 256 + threadIdx.x;
    int n = (int)(i % Npts);
    int o = (int)((i / Npts) % O);
    int b = (int)(i / ((size_t)Npts * O));
    size_t src = ((size_t)b * O + o) * Npts + n;
    float sc = scaleA[o];
    float y  = (sc >= 0.f) ? ymax[src] : ymin[src];
    float t  = fmaf(sc, y, shiftA[o]);
    float a  = (t >= 0.f) ? t : SLOPE * t;
    xc[((size_t)b * 512 + c0 + o) * Npts + n] = a;
    xcbf[((size_t)b * Npts + n) * 512 + c0 + o] = f2bf(a);
}

// conv5 BN stats: one block per channel
__global__ __launch_bounds__(256)
void conv5_stats_kernel(const float* __restrict__ y5,
                        const float* __restrict__ g, const float* __restrict__ bb,
                        float* __restrict__ scaleA, float* __restrict__ shiftA,
                        int E, int Npts, int Bn)
{
    __shared__ float ss[256], ss2[256];
    int e = blockIdx.x, t = threadIdx.x;
    float s = 0.f, s2 = 0.f;
    for (int b = 0; b < Bn; ++b) {
        const float* row = y5 + ((size_t)b * E + e) * Npts;
        for (int n = t; n < Npts; n += 256) {
            float v = row[n]; s += v; s2 = fmaf(v, v, s2);
        }
    }
    ss[t] = s; ss2[t] = s2; __syncthreads();
    for (int off = 128; off > 0; off >>= 1) {
        if (t < off) { ss[t] += ss[t + off]; ss2[t] += ss2[t + off]; }
        __syncthreads();
    }
    if (t == 0) {
        float inv = 1.f / (float)(Bn * Npts);
        float mean = ss[0] * inv;
        float var  = ss2[0] * inv - mean * mean;
        float sc = g[e] * rsqrtf(var + BN_EPS);
        scaleA[e] = sc;
        shiftA[e] = bb[e] - mean * sc;
    }
}

// activated max & mean over points -> h[b][e]=max, h[b][E+e]=mean
__global__ __launch_bounds__(256)
void pool_kernel(const float* __restrict__ y5,
                 const float* __restrict__ scaleA, const float* __restrict__ shiftA,
                 float* __restrict__ h, int E, int Npts)
{
    __shared__ float sm[256], sa[256];
    int e = blockIdx.x, b = blockIdx.y, t = threadIdx.x;
    float sc = scaleA[e], sh = shiftA[e];
    const float* row = y5 + ((size_t)b * E + e) * Npts;
    float mx = -3.4e38f, s = 0.f;
    for (int n = t; n < Npts; n += 256) {
        float tv = fmaf(sc, row[n], sh);
        float a  = (tv >= 0.f) ? tv : SLOPE * tv;
        mx = fmaxf(mx, a); s += a;
    }
    sm[t] = mx; sa[t] = s; __syncthreads();
    for (int off = 128; off > 0; off >>= 1) {
        if (t < off) { sm[t] = fmaxf(sm[t], sm[t + off]); sa[t] += sa[t + off]; }
        __syncthreads();
    }
    if (t == 0) {
        h[(size_t)b * 2 * E + e]     = sm[0];
        h[(size_t)b * 2 * E + E + e] = sa[0] / (float)Npts;
    }
}

// tiny FC: out[b][j] = in[b]·w[j] + bias
__global__ void fc_kernel(const float* __restrict__ in, const float* __restrict__ w,
                          const float* __restrict__ bias, float* __restrict__ out,
                          int Cin, int Cout, int Bn)
{
    int i = blockIdx.x * blockDim.x + threadIdx.x;
    if (i >= Bn * Cout) return;
    int b = i / Cout, j = i % Cout;
    float s = bias ? bias[j] : 0.f;
    const float* ir = in + (size_t)b * Cin;
    const float* wr = w + (size_t)j * Cin;
    for (int c = 0; c < Cin; ++c) s = fmaf(ir[c], wr[c], s);
    out[i] = s;
}

// BN over batch dim (Bn samples) + leaky relu
__global__ void bn_fc_kernel(const float* __restrict__ pre,
                             const float* __restrict__ g, const float* __restrict__ bb,
                             float* __restrict__ out, int F, int Bn)
{
    int j = blockIdx.x * blockDim.x + threadIdx.x;
    if (j >= F) return;
    float s = 0.f, s2 = 0.f;
    for (int b = 0; b < Bn; ++b) {
        float v = pre[(size_t)b * F + j]; s += v; s2 = fmaf(v, v, s2);
    }
    float inv = 1.f / (float)Bn;
    float mean = s * inv;
    float var  = s2 * inv - mean * mean;
    float sc = g[j] * rsqrtf(var + BN_EPS);
    float sh = bb[j] - mean * sc;
    for (int b = 0; b < Bn; ++b) {
        float t = fmaf(sc, pre[(size_t)b * F + j], sh);
        out[(size_t)b * F + j] = (t >= 0.f) ? t : SLOPE * t;
    }
}

// split weight [O][2C] into bf16 W_diff and (W_ctr - W_diff), zero-padded to Kp
__global__ void wtrans_kernel(const float* __restrict__ w,
                              unsigned short* __restrict__ wd,
                              unsigned short* __restrict__ wcd,
                              int O, int C, int Kp)
{
    int i = blockIdx.x * blockDim.x + threadIdx.x;
    if (i >= O * Kp) return;
    int o = i / Kp, kk = i % Kp;
    float d = 0.f, cd = 0.f;
    if (kk < C) {
        float wdv = w[(size_t)o * 2 * C + kk];
        float wcv = w[(size_t)o * 2 * C + C + kk];
        d = wdv; cd = wcv - wdv;
    }
    wd[i]  = f2bf(d);
    wcd[i] = f2bf(cd);
}

__global__ void f32_to_bf16_kernel(const float* __restrict__ src,
                                   unsigned short* __restrict__ dst, int n)
{
    int i = blockIdx.x * blockDim.x + threadIdx.x;
    if (i < n) dst[i] = f2bf(src[i]);
}

// input x [B][3][N] f32 -> bf16 point-major [B][N][32], zero-padded channels
__global__ void pad_input_kernel(const float* __restrict__ x,
                                 unsigned short* __restrict__ x0bf, int Npts, int Bn)
{
    int i = blockIdx.x * blockDim.x + threadIdx.x;
    int total = Bn * Npts * 32;
    if (i >= total) return;
    int c = i & 31;
    int n = (i >> 5) % Npts;
    int b = i / (32 * Npts);
    float v = (c < 3) ? x[((size_t)b * 3 + c) * Npts + n] : 0.f;
    x0bf[i] = f2bf(v);
}

// ---------------------------------------------------------------------------
extern "C" void kernel_launch(void* const* d_in, const int* in_sizes, int n_in,
                              void* d_out, int out_size, void* d_ws, size_t ws_size,
                              hipStream_t stream)
{
    (void)n_in; (void)out_size; (void)ws_size;
    const int Bn = 8;
    const int N  = in_sizes[0] / (Bn * 3);   // 2048
    const int E  = 1024;                     // embedding channels

    const float* x   = (const float*)d_in[0];
    const float* w1  = (const float*)d_in[1];
    const float* g1  = (const float*)d_in[2];
    const float* b1  = (const float*)d_in[3];
    const float* w2  = (const float*)d_in[4];
    const float* g2  = (const float*)d_in[5];
    const float* b2  = (const float*)d_in[6];
    const float* w3  = (const float*)d_in[7];
    const float* g3  = (const float*)d_in[8];
    const float* b3  = (const float*)d_in[9];
    const float* w4  = (const float*)d_in[10];
    const float* g4  = (const float*)d_in[11];
    const float* b4  = (const float*)d_in[12];
    const float* w5  = (const float*)d_in[13];
    const float* g5  = (const float*)d_in[14];
    const float* b5  = (const float*)d_in[15];
    const float* l1w = (const float*)d_in[16];
    const float* g6  = (const float*)d_in[17];
    const float* b6  = (const float*)d_in[18];
    const float* l2w = (const float*)d_in[19];
    const float* l2b = (const float*)d_in[20];
    const float* g7  = (const float*)d_in[21];
    const float* b7  = (const float*)d_in[22];
    const float* l3w = (const float*)d_in[23];
    const float* l3b = (const float*)d_in[24];
    // d_in[25] = k (always 20 in this model; hardcoded as KNN_K)

    // ---- workspace carve-up (256B aligned) ----
    char* p = (char*)d_ws;
    size_t off = 0;
    auto alloc = [&](size_t bytes) -> char* {
        char* r = p + off;
        off = (off + bytes + 255) & ~(size_t)255;
        return r;
    };
    float*          xc    = (float*)alloc((size_t)Bn * 512 * N * 4);
    unsigned short* xcbf  = (unsigned short*)alloc((size_t)Bn * N * 512 * 2);
    unsigned short* x0bf  = (unsigned short*)alloc((size_t)Bn * N * 32 * 2);
    int*            idx   = (int*)alloc((size_t)Bn * N * KNN_K * 4);
    // big region: u|v|ymax|ymin during edge convs, aliased as y5 for conv5
    float* big  = (float*)alloc((size_t)Bn * E * N * 4);
    float* u    = big;
    float* v    = big + (size_t)Bn * 256 * N;
    float* ymax = big + (size_t)2 * Bn * 256 * N;
    float* ymin = big + (size_t)3 * Bn * 256 * N;
    float* y5   = big;
    unsigned short* wd   = (unsigned short*)alloc(256 * 128 * 2);
    unsigned short* wcd  = (unsigned short*)alloc(256 * 128 * 2);
    unsigned short* w5bf = (unsigned short*)alloc((size_t)E * 512 * 2);
    float* statsBuf = (float*)alloc(2 * 1024 * 4);     // chS | chS2
    float* chS  = statsBuf;
    float* chS2 = statsBuf + 1024;
    float* scaleA = (float*)alloc(1024 * 4);
    float* shiftA = (float*)alloc(1024 * 4);
    float* h   = (float*)alloc((size_t)Bn * 2 * E * 4);
    float* h1p = (float*)alloc((size_t)Bn * 512 * 4);
    float* h1  = (float*)alloc((size_t)Bn * 512 * 4);
    float* h2p = (float*)alloc((size_t)Bn * 256 * 4);
    float* h2  = (float*)alloc((size_t)Bn * 256 * 4);

    // ---- input conversion (padded to 32 bf16 channels for WMMA K=32) ----
    {
        int tot = Bn * N * 32;
        pad_input_kernel<<<(tot + 255) / 256, 256, 0, stream>>>(x, x0bf, N, Bn);
    }

    // ---- edge conv layers ----
    struct LayerCfg {
        const float *w, *g, *b;
        const float* knn_src; long long knn_bstride; int C; int Kp;
        const unsigned short* bmat; int ldb; int bc0;
        int O; int c0_out;
    };
    LayerCfg layers[4] = {
        { w1, g1, b1, x,               (long long)3   * N, 3,   32,  x0bf, 32,  0,   64,  0   },
        { w2, g2, b2, xc + 0   * N,    (long long)512 * N, 64,  64,  xcbf, 512, 0,   64,  64  },
        { w3, g3, b3, xc + 64  * N,    (long long)512 * N, 64,  64,  xcbf, 512, 64,  128, 128 },
        { w4, g4, b4, xc + 128 * N,    (long long)512 * N, 128, 128, xcbf, 512, 128, 256, 256 },
    };

    const float invCnt = 1.f / (float)(Bn * N * KNN_K);
    for (int L = 0; L < 4; ++L) {
        const LayerCfg& cf = layers[L];

        knn_kernel<<<dim3(N / 64, Bn), 64, 0, stream>>>(
            cf.knn_src, cf.knn_bstride, cf.C, N, idx);

        wtrans_kernel<<<(cf.O * cf.Kp + 255) / 256, 256, 0, stream>>>(
            cf.w, wd, wcd, cf.O, cf.C, cf.Kp);

        long long bStrideB = (long long)N * cf.ldb;
        long long bStrideD = (long long)cf.O * N;
        dim3 gg(N / 64, cf.O / 16, Bn);
        gemm_bf16_wmma<<<gg, 32, 0, stream>>>(wcd, cf.Kp, cf.bmat, bStrideB,
                                              cf.ldb, cf.bc0, u, bStrideD,
                                              cf.O, N, cf.Kp);
        gemm_bf16_wmma<<<gg, 32, 0, stream>>>(wd, cf.Kp, cf.bmat, bStrideB,
                                              cf.ldb, cf.bc0, v, bStrideD,
                                              cf.O, N, cf.Kp);

        hipMemsetAsync(statsBuf, 0, 2 * 1024 * sizeof(float), stream);
        gather_stats_kernel<<<dim3(N / 32, cf.O / 8, Bn), dim3(32, 8), 0, stream>>>(
            u, v, idx, ymax, ymin, chS, chS2, cf.O, N);

        bn_prep_kernel<<<(cf.O + 63) / 64, 64, 0, stream>>>(
            chS, chS2, cf.g, cf.b, scaleA, shiftA, cf.O, invCnt);

        size_t tot = (size_t)Bn * cf.O * N;
        finalize_edge_kernel<<<(unsigned)(tot / 256), 256, 0, stream>>>(
            ymax, ymin, scaleA, shiftA, xc, xcbf, cf.O, cf.c0_out, N);
    }

    // ---- conv5: y5 = w5 · xc  (1024 x 512 x (B·N)) ----
    f32_to_bf16_kernel<<<(E * 512 + 255) / 256, 256, 0, stream>>>(w5, w5bf, E * 512);
    gemm_bf16_wmma<<<dim3(N / 64, E / 16, Bn), 32, 0, stream>>>(
        w5bf, 512, xcbf, (long long)N * 512, 512, 0,
        y5, (long long)E * N, E, N, 512);

    conv5_stats_kernel<<<E, 256, 0, stream>>>(y5, g5, b5, scaleA, shiftA, E, N, Bn);
    pool_kernel<<<dim3(E, Bn), 256, 0, stream>>>(y5, scaleA, shiftA, h, E, N);

    // ---- classifier head ----
    fc_kernel<<<(Bn * 512 + 255) / 256, 256, 0, stream>>>(h, l1w, nullptr, h1p, 2 * E, 512, Bn);
    bn_fc_kernel<<<(512 + 63) / 64, 64, 0, stream>>>(h1p, g6, b6, h1, 512, Bn);
    fc_kernel<<<(Bn * 256 + 255) / 256, 256, 0, stream>>>(h1, l2w, l2b, h2p, 512, 256, Bn);
    bn_fc_kernel<<<(256 + 63) / 64, 64, 0, stream>>>(h2p, g7, b7, h2, 256, Bn);
    fc_kernel<<<(Bn * 40 + 255) / 256, 256, 0, stream>>>(h2, l3w, l3b, (float*)d_out, 256, 40, Bn);
}
